// Graph_Attention_67181878444390
// MI455X (gfx1250) — matrix operand order repaired and checked by reading, not compiled, not run on hardware
//
#include <hip/hip_runtime.h>
#include <math.h>

// Problem constants
#define BB   4
#define BN   2048     // N
#define DD   128      // D
#define HH   4
#define EE   32768    // E
#define WW   8
#define OUTD 128
#define HD   512      // H*D

typedef __attribute__((ext_vector_type(16))) _Float16 v16h;
typedef __attribute__((ext_vector_type(8)))  float    v8f;

// -------------------- zero fill --------------------
__global__ void zero_kernel(float* __restrict__ p, long long count) {
    long long i = (long long)blockIdx.x * blockDim.x + threadIdx.x;
    long long stride = (long long)gridDim.x * blockDim.x;
    for (; i < count; i += stride) p[i] = 0.0f;
}

// -------------------- QKV projection via WMMA --------------------
// q[b,h,n,e] = sum_d X[b,n,d] * W[h,e,d] + bias[h,e]
// grid: (N/128, B*H, 3{q,k,v}), block 256 (8 waves, 16 rows each)
__global__ void __launch_bounds__(256) qkv_kernel(
    const float* __restrict__ X,
    const float* __restrict__ wq, const float* __restrict__ bq,
    const float* __restrict__ wk, const float* __restrict__ bk,
    const float* __restrict__ wv, const float* __restrict__ bv,
    float* __restrict__ Q, float* __restrict__ K, float* __restrict__ V)
{
    const int z = blockIdx.z;
    const float* W  = (z == 0) ? wq : (z == 1) ? wk : wv;
    const float* Bi = (z == 0) ? bq : (z == 1) ? bk : bv;
    float*       O  = (z == 0) ? Q  : (z == 1) ? K  : V;

    const int bh   = blockIdx.y;
    const int b    = bh / HH, h = bh % HH;
    const int wave = threadIdx.x >> 5;
    const int lane = threadIdx.x & 31;
    const int l    = lane & 15;
    const int hi   = (lane >> 4) & 1;

    const int m0 = blockIdx.x * 128 + wave * 16;
    const float* Xrow = X + ((size_t)b * BN + (m0 + l)) * DD;
    const float* Wh   = W + (size_t)h * DD * DD;
    const float* bias = Bi + h * DD;
    float* Obase = O + (size_t)bh * BN * DD;

    v8f acc[8];
    #pragma unroll
    for (int t = 0; t < 8; ++t)
        #pragma unroll
        for (int r = 0; r < 8; ++r) acc[t][r] = 0.0f;

    #pragma unroll
    for (int ks = 0; ks < 4; ++ks) {
        const int kbase = ks * 32;
        const int ka = kbase + (hi ? 8 : 0);    // A chunk 1 (8 contiguous K)
        const int kb = kbase + (hi ? 24 : 16);  // A chunk 2
        v16h a;
        #pragma unroll
        for (int t = 0; t < 8; ++t) a[t]     = (_Float16)Xrow[ka + t];
        #pragma unroll
        for (int t = 0; t < 8; ++t) a[8 + t] = (_Float16)Xrow[kb + t];

        const int kcol = kbase + (hi ? 8 : 0);  // B: two chunks of 8 K, matching A
        #pragma unroll
        for (int nt = 0; nt < 8; ++nt) {
            const float* wp = Wh + (size_t)(nt * 16 + l) * DD + kcol; // W[e, d..]
            v16h bf;
            #pragma unroll
            for (int t = 0; t < 16; ++t) bf[t] = (_Float16)wp[t < 8 ? t : t + 8];
            acc[nt] = __builtin_amdgcn_wmma_f32_16x16x32_f16(
                false, a, false, bf, (short)0, acc[nt], false, false);
        }
    }

    #pragma unroll
    for (int nt = 0; nt < 8; ++nt) {
        const int col = nt * 16 + l;
        const float bval = bias[col];
        #pragma unroll
        for (int r = 0; r < 8; ++r) {
            const int row = m0 + r + (hi ? 8 : 0);
            Obase[(size_t)row * DD + col] = acc[nt][r] + bval;
        }
    }
}

// -------------------- per-(b,h) column sum of V --------------------
__global__ void __launch_bounds__(128) vtot_kernel(
    const float* __restrict__ V, float* __restrict__ Vtot)
{
    const int bh = blockIdx.x;
    const int d  = threadIdx.x;
    const float* vb = V + (size_t)bh * BN * DD;
    float s = 0.0f;
    for (int n = 0; n < BN; ++n) s += vb[(size_t)n * DD + d];
    Vtot[bh * DD + d] = s;
}

// -------------------- CSR build: count edges per target node --------------------
__global__ void __launch_bounds__(256) count_kernel(
    const int* __restrict__ idx, int* __restrict__ cnt)
{
    const int t = blockIdx.x * 256 + threadIdx.x;   // B*E threads
    const int b = t / EE, e = t % EE;
    const int i = idx[(size_t)b * 2 * EE + e];
    atomicAdd(&cnt[b * BN + i], 1);
}

// -------------------- CSR build: serial exclusive scan per batch --------------------
__global__ void scan_kernel(const int* __restrict__ cnt,
                            int* __restrict__ offs, int* __restrict__ cursor)
{
    const int b = threadIdx.x;
    if (b >= BB) return;
    int run = 0;
    for (int n = 0; n < BN; ++n) {
        const int c = cnt[b * BN + n];
        offs[b * BN + n]   = run;
        cursor[b * BN + n] = run;
        run += c;
    }
}

// -------------------- CSR build: bucket fill --------------------
__global__ void __launch_bounds__(256) fill_kernel(
    const int* __restrict__ idx, int* __restrict__ cursor,
    int* __restrict__ edgelist)
{
    const int t = blockIdx.x * 256 + threadIdx.x;   // B*E threads
    const int b = t / EE, e = t % EE;
    const int i = idx[(size_t)b * 2 * EE + e];
    const int slot = atomicAdd(&cursor[b * BN + i], 1);
    edgelist[b * EE + slot] = e;
}

// -------------------- per-edge softmax weights (all heads, no atomics) --------------------
// one wave per (b,e); grid = B*E/8 blocks of 256
__global__ void __launch_bounds__(256) wgt_kernel(
    const float* __restrict__ Q, const float* __restrict__ K,
    const int* __restrict__ idx, const float* __restrict__ attrs,
    float* __restrict__ Wgt)
{
    const int wave = threadIdx.x >> 5;
    const int lane = threadIdx.x & 31;
    const int task = blockIdx.x * 8 + wave;
    const int b = task / EE, e = task % EE;

    const int i = idx[(size_t)b * 2 * EE + e];
    const int j = idx[(size_t)b * 2 * EE + EE + e];

    float wsum = 0.0f;
    const float* ap = attrs + ((size_t)b * EE + e) * WW;
    #pragma unroll
    for (int t = 0; t < WW; ++t) wsum += ap[t];
    const float scale = 0.022097086912079612f * wsum;  // wsum / sqrt(2048)

    #pragma unroll
    for (int h = 0; h < HH; ++h) {
        const int bh = b * HH + h;
        const float* qrow = Q + ((size_t)bh * BN + i) * DD;
        const float* krow = K + ((size_t)bh * BN + j) * DD;
        float part = 0.0f;
        #pragma unroll
        for (int c = 0; c < 4; ++c) {
            const int d = lane + 32 * c;
            part += qrow[d] * krow[d];
        }
        #pragma unroll
        for (int off = 16; off > 0; off >>= 1)
            part += __shfl_xor(part, off, 32);
        if (lane == 0)
            Wgt[(size_t)bh * EE + e] = expf(part * scale) - 1.0f;
    }
}

// -------------------- gather: sparse numerator + Z per (b,h,n), no atomics --------------------
// one wave per (b,h,n); grid = B*H*N/8 blocks of 256
__global__ void __launch_bounds__(256) gather_kernel(
    const float* __restrict__ V, const int* __restrict__ idx,
    const float* __restrict__ Wgt,
    const int* __restrict__ offs, const int* __restrict__ cnt,
    const int* __restrict__ edgelist,
    float* __restrict__ Acc, float* __restrict__ Znum)
{
    const int wave = threadIdx.x >> 5;
    const int lane = threadIdx.x & 31;
    const int task = blockIdx.x * 8 + wave;   // 0 .. B*H*N-1
    const int bh = task / BN;
    const int n  = task % BN;
    const int b  = bh / HH;

    const int base  = b * BN + n;
    const int start = offs[base];
    const int c     = cnt[base];

    const int*   el   = edgelist + (size_t)b * EE;
    const int*   jrow = idx + (size_t)b * 2 * EE + EE;
    const float* Wh   = Wgt + (size_t)bh * EE;
    const float* Vb   = V + (size_t)bh * BN * DD;

    float accv[4] = {0.0f, 0.0f, 0.0f, 0.0f};
    float z = 0.0f;
    for (int s = 0; s < c; ++s) {
        const int   e = el[start + s];
        const int   j = jrow[e];
        const float w = Wh[e];
        z += w;
        const float* vr = Vb + (size_t)j * DD;
        #pragma unroll
        for (int cc = 0; cc < 4; ++cc)
            accv[cc] += w * vr[lane + 32 * cc];
    }

    float* arow = Acc + ((size_t)bh * BN + n) * DD;
    #pragma unroll
    for (int cc = 0; cc < 4; ++cc)
        arow[lane + 32 * cc] = accv[cc];
    if (lane == 0) Znum[(size_t)bh * BN + n] = z;
}

// -------------------- fused finalize + output projection (WMMA) --------------------
// o[b,n,c] = (Vtot[b,h,d] + Acc[b,h,n,d]) / (N + Znum[b,h,n]),  c = h*128+d
// out[b,n,o] = sum_c o[b,n,c] * out_w[o,c] + out_b[o]
__global__ void __launch_bounds__(256) out_kernel(
    const float* __restrict__ Acc, const float* __restrict__ Znum,
    const float* __restrict__ Vtot,
    const float* __restrict__ out_w, const float* __restrict__ out_b,
    float* __restrict__ out)
{
    const int wave = threadIdx.x >> 5;
    const int lane = threadIdx.x & 31;
    const int l    = lane & 15;
    const int hi   = (lane >> 4) & 1;
    const int m0   = blockIdx.x * 128 + wave * 16;
    const int rn   = m0 + l;          // global row in [0, B*N)
    const int b    = rn / BN;
    const int n    = rn % BN;

    v8f acc[8];
    #pragma unroll
    for (int t = 0; t < 8; ++t)
        #pragma unroll
        for (int r = 0; r < 8; ++r) acc[t][r] = 0.0f;

    for (int ks = 0; ks < 16; ++ks) {
        const int kbase = ks * 32;
        const int h     = kbase >> 7;          // constant within this k-step
        const int bhn   = (b * HH + h) * BN + n;
        const float recip = 1.0f / ((float)BN + Znum[bhn]);
        const float* arow = Acc  + (size_t)bhn * DD;
        const float* vt   = Vtot + (size_t)(b * HH + h) * DD;
        const int c0 = (kbase & 127) + (hi ? 8 : 0);
        const int c1 = (kbase & 127) + (hi ? 24 : 16);
        v16h a;
        #pragma unroll
        for (int t = 0; t < 8; ++t)
            a[t]     = (_Float16)((vt[c0 + t] + arow[c0 + t]) * recip);
        #pragma unroll
        for (int t = 0; t < 8; ++t)
            a[8 + t] = (_Float16)((vt[c1 + t] + arow[c1 + t]) * recip);

        const int kcol = kbase + (hi ? 8 : 0);
        #pragma unroll
        for (int nt = 0; nt < 8; ++nt) {
            const float* wp = out_w + (size_t)(nt * 16 + l) * HD + kcol;
            v16h bf;
            #pragma unroll
            for (int t = 0; t < 16; ++t) bf[t] = (_Float16)wp[t < 8 ? t : t + 8];
            acc[nt] = __builtin_amdgcn_wmma_f32_16x16x32_f16(
                false, a, false, bf, (short)0, acc[nt], false, false);
        }
    }

    #pragma unroll
    for (int nt = 0; nt < 8; ++nt) {
        const int col = nt * 16 + l;
        const float bval = out_b[col];
        #pragma unroll
        for (int r = 0; r < 8; ++r) {
            const int row = m0 + r + (hi ? 8 : 0);
            out[(size_t)row * OUTD + col] = acc[nt][r] + bval;
        }
    }
}

// -------------------- launcher --------------------
extern "C" void kernel_launch(void* const* d_in, const int* in_sizes, int n_in,
                              void* d_out, int out_size, void* d_ws, size_t ws_size,
                              hipStream_t stream)
{
    (void)in_sizes; (void)n_in; (void)out_size; (void)ws_size;
    const float* X   = (const float*)d_in[0];
    const int*   idx = (const int*)  d_in[1];
    const float* at  = (const float*)d_in[2];
    const float* wq  = (const float*)d_in[3];
    const float* bq  = (const float*)d_in[4];
    const float* wk  = (const float*)d_in[5];
    const float* bk  = (const float*)d_in[6];
    const float* wv  = (const float*)d_in[7];
    const float* bv  = (const float*)d_in[8];
    const float* ow  = (const float*)d_in[9];
    const float* ob  = (const float*)d_in[10];
    float* out = (float*)d_out;

    float* ws = (float*)d_ws;
    const size_t QKV = (size_t)BB * HH * BN * DD;   // 4,194,304 floats
    float* Q    = ws;
    float* K    = Q + QKV;
    float* V    = K + QKV;
    float* Acc  = V + QKV;
    float* Znum = Acc + QKV;                         // B*H*N
    float* Vtot = Znum + (size_t)BB * HH * BN;       // B*H*D
    float* Wgt  = Vtot + (size_t)BB * HH * DD;       // B*H*E
    int*   cnt      = (int*)(Wgt + (size_t)BB * HH * EE);
    int*   offs     = cnt    + (size_t)BB * BN;
    int*   cursor   = offs   + (size_t)BB * BN;
    int*   edgelist = cursor + (size_t)BB * BN;      // B*E
    // total ws ~= 67 MB

    // zero the per-node edge counts (ints; 0.0f bit pattern == 0)
    zero_kernel<<<32, 256, 0, stream>>>((float*)cnt, (long long)(BB * BN));

    // QKV projections (WMMA)
    dim3 g1(BN / 128, BB * HH, 3);
    qkv_kernel<<<g1, 256, 0, stream>>>(X, wq, bq, wk, bk, wv, bv, Q, K, V);

    // column sums of V per (b,h)
    vtot_kernel<<<BB * HH, 128, 0, stream>>>(V, Vtot);

    // build per-batch CSR of edges keyed by target node i
    count_kernel<<<(BB * EE) / 256, 256, 0, stream>>>(idx, cnt);
    scan_kernel<<<1, 32, 0, stream>>>(cnt, offs, cursor);
    fill_kernel<<<(BB * EE) / 256, 256, 0, stream>>>(idx, cursor, edgelist);

    // per-edge softmax weights for all heads
    wgt_kernel<<<(BB * EE) / 8, 256, 0, stream>>>(Q, K, idx, at, Wgt);

    // gather sparse numerator + Z per (b,h,n) — no atomics
    gather_kernel<<<(BB * HH * BN) / 8, 256, 0, stream>>>(
        V, idx, Wgt, offs, cnt, edgelist, Acc, Znum);

    // finalize + output projection (WMMA)
    out_kernel<<<(BB * BN) / 128, 256, 0, stream>>>(Acc, Znum, Vtot, ow, ob, out);
}
